// GCN_E2_decline_L3_sqr_clf1_selu_multi_83786222010498
// MI455X (gfx1250) — compile-verified, matrix operand order checked by reading, not run on hardware
//
#include <hip/hip_runtime.h>
#include <hip/hip_bf16.h>

// ---------------------------------------------------------------------------
// CDNA5 (gfx1250) GCN forward: 3x (GEMM -> edge SpMM -> SELU+bias) -> 2 heads
// Main GEMM uses v_wmma_f32_16x16x32_bf16 (bf16 A/B, f32 accum), wave32.
// ---------------------------------------------------------------------------

typedef __attribute__((ext_vector_type(16))) __bf16 v16bf;
typedef __attribute__((ext_vector_type(8)))  float  v8f;

#define HID 32

// ---------------------------------------------------------------------------
// Pack a row-major [K x 32] f32 weight matrix into bf16 WMMA B-fragment order.
// Fragment order: for each k-block kb (32 K's), col-group g (16 cols), lane,
// 16 contiguous halfs matching the 16-bit B-matrix VGPR layout:
//   kbase = (lane<16)?0:8 ; i<8 -> K=kb*32+kbase+i ; i>=8 -> K=kb*32+16+kbase+(i-8)
//   N = g*16 + (lane&15)
// ---------------------------------------------------------------------------
__global__ __launch_bounds__(256) void pack_w_bf16(const float* __restrict__ W,
                                                   __bf16* __restrict__ Wp,
                                                   int total) {
    int idx = blockIdx.x * blockDim.x + threadIdx.x;
    if (idx >= total) return;
    int i    = idx & 15;
    int lane = (idx >> 4) & 31;
    int g    = (idx >> 9) & 1;
    int kb   = idx >> 10;
    int kbase = (lane < 16) ? 0 : 8;
    int K = kb * 32 + ((i < 8) ? (kbase + i) : (16 + kbase + (i - 8)));
    int N = g * 16 + (lane & 15);
    Wp[idx] = (__bf16)W[K * HID + N];
}

// ---------------------------------------------------------------------------
// GEMM: Out[M x 32] = X[M x K] @ W[K x 32], K % 32 == 0, M % 16 == 0.
// One wave computes one 16-row tile across all 32 output columns
// (two 16x16 C fragments). A is converted f32 -> bf16 in registers; B is
// pre-packed in fragment order (contiguous 32B v16bf load per fragment).
// ---------------------------------------------------------------------------
__global__ __launch_bounds__(256) void gemm_wmma_bf16(const float* __restrict__ X,
                                                      const __bf16* __restrict__ Wp,
                                                      float* __restrict__ Out,
                                                      int M, int K) {
    const int wave = (blockIdx.x * blockDim.x + threadIdx.x) >> 5;
    const int lane = threadIdx.x & 31;
    const int nRowTiles = M >> 4;
    if (wave >= nRowTiles) return;   // uniform per-wave guard: EXEC stays all-1s

    const int row   = (wave << 4) + (lane & 15);
    const int kbase = (lane < 16) ? 0 : 8;
    const float* xr = X + (size_t)row * K + kbase;

    v8f c0 = {};
    v8f c1 = {};
    const int kblocks = K >> 5;

    for (int kb = 0; kb < kblocks; ++kb) {
        const float* p = xr + (kb << 5);
        // Two contiguous 8-float runs per lane: K = kbase..+7 and 16+kbase..+7
        float4 f0 = *(const float4*)(p);
        float4 f1 = *(const float4*)(p + 4);
        float4 f2 = *(const float4*)(p + 16);
        float4 f3 = *(const float4*)(p + 20);
        __builtin_prefetch(p + 32, 0, 0);   // next k-block -> global_prefetch

        v16bf a;
        a[0]  = (__bf16)f0.x; a[1]  = (__bf16)f0.y; a[2]  = (__bf16)f0.z; a[3]  = (__bf16)f0.w;
        a[4]  = (__bf16)f1.x; a[5]  = (__bf16)f1.y; a[6]  = (__bf16)f1.z; a[7]  = (__bf16)f1.w;
        a[8]  = (__bf16)f2.x; a[9]  = (__bf16)f2.y; a[10] = (__bf16)f2.z; a[11] = (__bf16)f2.w;
        a[12] = (__bf16)f3.x; a[13] = (__bf16)f3.y; a[14] = (__bf16)f3.z; a[15] = (__bf16)f3.w;

        const v16bf b0 = *(const v16bf*)(Wp + ((size_t)(kb * 2 + 0) * 32 + lane) * 16);
        const v16bf b1 = *(const v16bf*)(Wp + ((size_t)(kb * 2 + 1) * 32 + lane) * 16);

        c0 = __builtin_amdgcn_wmma_f32_16x16x32_bf16(false, a, false, b0,
                                                     (short)0, c0, false, false);
        c1 = __builtin_amdgcn_wmma_f32_16x16x32_bf16(false, a, false, b1,
                                                     (short)0, c1, false, false);
    }

    // C/D layout: lane holds column (lane&15); VGPR r holds row r (+8 for hi half)
    const int n0    = lane & 15;
    const int mbase = (wave << 4) + (((lane >> 4) & 1) << 3);
    float* o = Out + (size_t)mbase * HID;
    #pragma unroll
    for (int r = 0; r < 8; ++r) {
        o[(size_t)r * HID + n0]      = c0[r];
        o[(size_t)r * HID + 16 + n0] = c1[r];
    }
}

// ---------------------------------------------------------------------------
// Zero-fill f32 buffer
// ---------------------------------------------------------------------------
__global__ __launch_bounds__(256) void zero_f32(float* __restrict__ A, int total) {
    int idx = blockIdx.x * blockDim.x + threadIdx.x;
    if (idx < total) A[idx] = 0.0f;
}

// ---------------------------------------------------------------------------
// Edge SpMM scatter: agg[row] += val * support[col], 32 feats per edge.
// One lane per (edge, feature); agg (6.4MB) is L2-resident -> cheap atomics.
// ---------------------------------------------------------------------------
__global__ __launch_bounds__(256) void spmm_scatter(const int* __restrict__ rows,
                                                    const int* __restrict__ cols,
                                                    const float* __restrict__ vals,
                                                    const float* __restrict__ S,
                                                    float* __restrict__ A, int E) {
    long long t = (long long)blockIdx.x * blockDim.x + threadIdx.x;
    int e = (int)(t >> 5);
    int f = (int)(t & 31);
    if (e >= E) return;
    int   r = rows[e];
    int   c = cols[e];
    float v = vals[e];
    atomicAdd(&A[(size_t)r * HID + f], v * S[(size_t)c * HID + f]);
}

// ---------------------------------------------------------------------------
// h = selu(agg) + b   (bias added AFTER activation, per reference)
// ---------------------------------------------------------------------------
__global__ __launch_bounds__(256) void selu_bias(const float* __restrict__ A,
                                                 const float* __restrict__ b,
                                                 float* __restrict__ H, int total) {
    int idx = blockIdx.x * blockDim.x + threadIdx.x;
    if (idx >= total) return;
    const float scale = 1.0507009873554805f;
    const float alpha = 1.6732632423543772f;
    float x = A[idx];
    float s = (x > 0.0f) ? (scale * x) : (scale * alpha * (__expf(x) - 1.0f));
    H[idx] = s + b[idx & (HID - 1)];
}

// ---------------------------------------------------------------------------
// Classifier heads: out0 = h @ w0^T + b0 (Nx2), out1 = h @ w1^T + b1 (Nx3)
// w stored torch-style [out, in]. Tiny -> one thread per node.
// ---------------------------------------------------------------------------
__global__ __launch_bounds__(256) void clf_heads(const float* __restrict__ H,
                                                 const float* __restrict__ w0,
                                                 const float* __restrict__ b0,
                                                 const float* __restrict__ w1,
                                                 const float* __restrict__ b1,
                                                 float* __restrict__ out, int N) {
    int n = blockIdx.x * blockDim.x + threadIdx.x;
    if (n >= N) return;
    const float* h = H + (size_t)n * HID;
    float a0 = b0[0], a1 = b0[1], a2 = b1[0], a3 = b1[1], a4 = b1[2];
    #pragma unroll
    for (int k = 0; k < HID; ++k) {
        float hv = h[k];
        a0 += hv * w0[0 * HID + k];
        a1 += hv * w0[1 * HID + k];
        a2 += hv * w1[0 * HID + k];
        a3 += hv * w1[1 * HID + k];
        a4 += hv * w1[2 * HID + k];
    }
    out[(size_t)n * 2 + 0] = a0;
    out[(size_t)n * 2 + 1] = a1;
    float* o1 = out + (size_t)N * 2;
    o1[(size_t)n * 3 + 0] = a2;
    o1[(size_t)n * 3 + 1] = a3;
    o1[(size_t)n * 3 + 2] = a4;
}

// ---------------------------------------------------------------------------
extern "C" void kernel_launch(void* const* d_in, const int* in_sizes, int n_in,
                              void* d_out, int out_size, void* d_ws, size_t ws_size,
                              hipStream_t stream) {
    const float* x        = (const float*)d_in[0];
    const int*   adj_rows = (const int*)d_in[1];
    const int*   adj_cols = (const int*)d_in[2];
    const float* adj_vals = (const float*)d_in[3];
    const float* w1  = (const float*)d_in[4];
    const float* b1  = (const float*)d_in[5];
    const float* w2  = (const float*)d_in[6];
    const float* b2  = (const float*)d_in[7];
    const float* w3  = (const float*)d_in[8];
    const float* b3  = (const float*)d_in[9];
    const float* cw0 = (const float*)d_in[10];
    const float* cb0 = (const float*)d_in[11];
    const float* cw1 = (const float*)d_in[12];
    const float* cb1 = (const float*)d_in[13];

    const int K1 = 1024;                    // IN_DIM
    const int N  = in_sizes[0] / K1;        // 50000 nodes (multiple of 16)
    const int E  = in_sizes[1];             // 1.6M edges

    // Workspace layout (all 256B-aligned by construction)
    const size_t nh = (size_t)N * HID;
    float*  support = (float*)d_ws;
    float*  agg     = support + nh;
    float*  h       = agg + nh;
    __bf16* w1p     = (__bf16*)(h + nh);               // (K1/32)*1024 halfs
    __bf16* w2p     = w1p + (size_t)(K1 / 32) * 1024;  // 1024 halfs
    __bf16* w3p     = w2p + 1024;

    const int elems      = (int)nh;
    const int eBlocks    = (elems + 255) / 256;
    const int rowTiles   = N / 16;
    const int gemmBlocks = (rowTiles + 7) / 8;          // 8 waves / 256-thread block
    const int spmmBlocks = (int)(((long long)E * 32 + 255) / 256);
    const int packW1     = (K1 / 32) * 1024;

    // Pack weights into bf16 WMMA fragment order
    pack_w_bf16<<<(packW1 + 255) / 256, 256, 0, stream>>>(w1, w1p, packW1);
    pack_w_bf16<<<4, 256, 0, stream>>>(w2, w2p, 1024);
    pack_w_bf16<<<4, 256, 0, stream>>>(w3, w3p, 1024);

    // Layer 1: big GEMM (memory-bound on x: ~205MB), then SpMM + SELU+bias
    gemm_wmma_bf16<<<gemmBlocks, 256, 0, stream>>>(x, w1p, support, N, K1);
    zero_f32<<<eBlocks, 256, 0, stream>>>(agg, elems);
    spmm_scatter<<<spmmBlocks, 256, 0, stream>>>(adj_rows, adj_cols, adj_vals,
                                                 support, agg, E);
    selu_bias<<<eBlocks, 256, 0, stream>>>(agg, b1, h, elems);

    // Layer 2
    gemm_wmma_bf16<<<gemmBlocks, 256, 0, stream>>>(h, w2p, support, N, HID);
    zero_f32<<<eBlocks, 256, 0, stream>>>(agg, elems);
    spmm_scatter<<<spmmBlocks, 256, 0, stream>>>(adj_rows, adj_cols, adj_vals,
                                                 support, agg, E);
    selu_bias<<<eBlocks, 256, 0, stream>>>(agg, b2, h, elems);

    // Layer 3
    gemm_wmma_bf16<<<gemmBlocks, 256, 0, stream>>>(h, w3p, support, N, HID);
    zero_f32<<<eBlocks, 256, 0, stream>>>(agg, elems);
    spmm_scatter<<<spmmBlocks, 256, 0, stream>>>(adj_rows, adj_cols, adj_vals,
                                                 support, agg, E);
    selu_bias<<<eBlocks, 256, 0, stream>>>(agg, b3, h, elems);

    // Heads
    clf_heads<<<(N + 255) / 256, 256, 0, stream>>>(h, cw0, cb0, cw1, cb1,
                                                   (float*)d_out, N);
}
